// SSVDModel_63823214018579
// MI455X (gfx1250) — compile-verified, matrix operand order checked by reading, not run on hardware
//
#include <hip/hip_runtime.h>

// ---------------------------------------------------------------------------
// Problem constants (from reference)
// ---------------------------------------------------------------------------
constexpr int BB   = 256;   // batch of matrices
constexpr int HH   = 128;   // matrix rows
constexpr int WW   = 128;   // matrix cols
constexpr int DD   = 29;    // obs channels
constexpr int KK   = 32;    // k
constexpr int AOUT = 256;   // output features per matrix
constexpr int KDIM = HH * (WW - KK);   // 128*96 = 12288
constexpr int CP   = 129;   // padded LDS column stride (bank-conflict avoidance)
constexpr int RP   = 33;    // padded stride for 32-wide r buffers

typedef __attribute__((ext_vector_type(16))) __bf16        v16bf;
typedef __attribute__((ext_vector_type(8)))  float         v8f;
typedef __attribute__((ext_vector_type(4)))  unsigned int  v4u;
typedef __attribute__((ext_vector_type(8)))  int           v8i;
typedef __attribute__((ext_vector_type(4)))  int           v4i;

union Frag { v4u u2[2]; v16bf v; };

#if defined(__has_builtin)
# if __has_builtin(__builtin_amdgcn_tensor_load_to_lds) && \
     __has_builtin(__builtin_amdgcn_s_wait_tensorcnt)
#  define HAVE_TDM 1
# endif
#endif
#ifndef HAVE_TDM
# define HAVE_TDM 0
#endif

__device__ __forceinline__ unsigned short f32_to_bf16(float f) {
  unsigned int u = __float_as_uint(f);
  u += 0x7FFFu + ((u >> 16) & 1u);     // round-to-nearest-even
  return (unsigned short)(u >> 16);
}

// ---------------------------------------------------------------------------
// Composite affine coefficients of the conv cascade (6 -> 4 -> 3 -> 2 -> 1)
// ---------------------------------------------------------------------------
__device__ float conv_chain6(const float x[6], const float* w1, float b1,
                             const float* w2, float b2) {
  float t4[4], t3[3], t2[2];
  #pragma unroll
  for (int t = 0; t < 4; ++t) {        // k=4, s=2, p=2 : 6 -> 4
    float a = b1;
    #pragma unroll
    for (int kk = 0; kk < 4; ++kk) {
      int i = 2 * t + kk - 2;
      a += w1[kk] * ((i >= 0 && i < 6) ? x[i] : 0.f);
    }
    t4[t] = a;
  }
  #pragma unroll
  for (int t = 0; t < 3; ++t) {        // 4 -> 3
    float a = b1;
    #pragma unroll
    for (int kk = 0; kk < 4; ++kk) {
      int i = 2 * t + kk - 2;
      a += w1[kk] * ((i >= 0 && i < 4) ? t4[i] : 0.f);
    }
    t3[t] = a;
  }
  #pragma unroll
  for (int t = 0; t < 2; ++t) {        // 3 -> 2
    float a = b1;
    #pragma unroll
    for (int kk = 0; kk < 4; ++kk) {
      int i = 2 * t + kk - 2;
      a += w1[kk] * ((i >= 0 && i < 3) ? t3[i] : 0.f);
    }
    t2[t] = a;
  }
  return b2 + w2[0] * t2[0] + w2[1] * t2[1];   // k=2, s=1, p=0 : 2 -> 1
}

// ---------------------------------------------------------------------------
// Kernel 0: convert weightsO to bf16 (L2-resident thereafter)
// ---------------------------------------------------------------------------
__global__ __launch_bounds__(256) void cvt_bf16_kernel(
    const float* __restrict__ in, unsigned short* __restrict__ out, int n) {
  int i = blockIdx.x * 256 + threadIdx.x;
  if (i < n) out[i] = f32_to_bf16(in[i]);
}

// ---------------------------------------------------------------------------
// Kernel 1: obs -> mats   (fused chunk-weighting + conv cascade, pure affine)
// ---------------------------------------------------------------------------
__global__ __launch_bounds__(256) void build_mats_kernel(
    const float* __restrict__ obs, const float* __restrict__ c1w,
    const float* __restrict__ c1b, const float* __restrict__ c2w,
    const float* __restrict__ c2b, float* __restrict__ mats) {
  const int FS[6] = {5, 5, 3, 8, 6, 2};
  float w1[4] = {c1w[0], c1w[1], c1w[2], c1w[3]};
  float w2[2] = {c2w[0], c2w[1]};
  float b1 = c1b[0], b2 = c2b[0];

  float z[6] = {0.f, 0.f, 0.f, 0.f, 0.f, 0.f};
  float c0 = conv_chain6(z, w1, b1, w2, b2);
  float cw[DD];
  int p = 0;
  #pragma unroll
  for (int j = 0; j < 6; ++j) {
    z[j] = 1.f;
    float coef = conv_chain6(z, w1, b1, w2, b2) - c0;
    z[j] = 0.f;
    for (int o = 0; o < FS[j]; ++o) cw[p++] = coef * (float)o;
  }

  const long long NP = (long long)BB * HH * WW;
  for (long long pix = (long long)blockIdx.x * 256 + threadIdx.x; pix < NP;
       pix += (long long)gridDim.x * 256) {
    const float* o = obs + pix * DD;
    float acc = c0;
    #pragma unroll
    for (int d = 0; d < DD; ++d) acc += cw[d] * o[d];
    mats[pix] = acc;
  }
}

// ---------------------------------------------------------------------------
// Kernel 2: per-matrix one-sided Jacobi SVD in LDS + relu-MLP chain -> rflat
// One workgroup (8 wave32) per 128x128 matrix; ~164 KB LDS (CDNA5: 320KB/WGP)
//
// The matrix tile is DMA'd by the Tensor Data Mover: one TENSOR_LOAD_TO_LDS
// with pad_interval=128 dwords / pad_amount=1 dword lands m row-major in LDS
// with stride CP=129.  Viewed column-major that LDS image is B = m^T, so
// Jacobi on B yields  U_m = accumulated rotations (Vc)  and
// V_m = normalized B-columns (Ac) — roles simply swap downstream.
// ---------------------------------------------------------------------------
__global__ __launch_bounds__(256) void svd_mlp_kernel(
    const float* __restrict__ mats, const float* __restrict__ w1,
    const float* __restrict__ w2, unsigned short* __restrict__ rflat) {
  extern __shared__ float smem[];
  float* Ac   = smem;                 // 128*CP  B = m^T column-major: Ac[c*CP+r]
  float* Vc   = Ac + 128 * CP;        // 128*CP  rotation accumulator (-> U_m)
  float* sv   = Vc + 128 * CP;        // 128 singular values
  float* sinv = sv + 128;             // 128 reciprocals (sorted order)
  int*   idx  = (int*)(sinv + 128);   // 128 sort permutation (descending)
  float* r1   = (float*)(idx + 128);  // 128*RP
  float* r2   = r1 + 128 * RP;        // 128*RP

  const int tid = threadIdx.x;
  const int mat = blockIdx.x;
  const float* M = mats + (size_t)mat * (HH * WW);

#if HAVE_TDM
  if (tid == 0) {
    unsigned long long ga = (unsigned long long)(uintptr_t)M;
    unsigned lds_off = (unsigned)(uintptr_t)(void*)Ac;   // low 32 bits = LDS offset
    v4u g0;
    g0[0] = 1u;                                          // count=1, user mode
    g0[1] = lds_off;                                     // lds_addr
    g0[2] = (unsigned)(ga & 0xFFFFFFFFu);                // global_addr[31:0]
    g0[3] = (unsigned)((ga >> 32) & 0x01FFFFFFu) | (2u << 30);  // addr[56:32]|type=2
    v8i g1;
    g1[0] = (2 << 16) | (1 << 20) | (6 << 22);  // data_size=4B, pad_en, every 128dw, +1dw
    g1[1] = (int)(128u << 16);                  // tensor_dim0[15:0]=128 (bits 63:48)
    g1[2] = (int)(128u << 16);                  // dim0[31:16]=0 | tensor_dim1[15:0]=128
    g1[3] = (int)(128u << 16);                  // dim1[31:16]=0 | tile_dim0=128
    g1[4] = 128;                                // tile_dim1=128, tile_dim2=0
    g1[5] = 128;                                // tensor_dim0_stride[31:0]=128
    g1[6] = (int)(0x4000u << 16);               // stride hi=0 | dim1_stride[15:0]=16384
    g1[7] = 0;                                  // dim1_stride[47:16]=0
    v4i gz = {0, 0, 0, 0};
#if defined(__clang_major__) && (__clang_major__ >= 23)
    v8i gz8 = {0, 0, 0, 0, 0, 0, 0, 0};
    __builtin_amdgcn_tensor_load_to_lds(g0, g1, gz, gz, gz8, 0);
#else
    __builtin_amdgcn_tensor_load_to_lds(g0, g1, gz, gz, 0);
#endif
  }
  // V = I while the TDM streams the tile in
  for (int e = tid; e < HH * WW; e += 256) {
    int r = e >> 7, c = e & 127;
    Vc[c * CP + r] = (r == c) ? 1.f : 0.f;
  }
  if (tid == 0) __builtin_amdgcn_s_wait_tensorcnt(0);
  __syncthreads();
#else
  for (int e = tid; e < HH * WW; e += 256) {
    int r = e >> 7, c = e & 127;
    Ac[r * CP + c] = M[e];                 // row-major image == m^T column-major
    Vc[c * CP + r] = (r == c) ? 1.f : 0.f;
  }
  __syncthreads();
#endif

  // One-sided Hestenes-Jacobi on B: 64 disjoint pairs/round, 4 lanes per pair
  const int pairId = tid >> 2;
  const int sub    = tid & 3;
  for (int sweep = 0; sweep < 8; ++sweep) {
    for (int rnd = 0; rnd < 127; ++rnd) {
      int p, q;
      if (pairId == 0) { p = 127; q = rnd % 127; }
      else {
        p = (rnd + pairId) % 127;
        q = (rnd - pairId + 127) % 127;
      }
      float* Ap = Ac + p * CP + sub * 32;
      float* Aq = Ac + q * CP + sub * 32;
      float app = 0.f, aqq = 0.f, apq = 0.f;
      #pragma unroll 8
      for (int r = 0; r < 32; ++r) {
        float x = Ap[r], y = Aq[r];
        app = fmaf(x, x, app);
        aqq = fmaf(y, y, aqq);
        apq = fmaf(x, y, apq);
      }
      app += __shfl_xor(app, 1); aqq += __shfl_xor(aqq, 1); apq += __shfl_xor(apq, 1);
      app += __shfl_xor(app, 2); aqq += __shfl_xor(aqq, 2); apq += __shfl_xor(apq, 2);
      if (apq * apq > 1e-24f * app * aqq) {
        float zeta = (aqq - app) / (2.f * apq);
        float t    = copysignf(1.f, zeta) / (fabsf(zeta) + sqrtf(fmaf(zeta, zeta, 1.f)));
        float cth  = rsqrtf(fmaf(t, t, 1.f));
        float sth  = cth * t;
        float* Vp = Vc + p * CP + sub * 32;
        float* Vq = Vc + q * CP + sub * 32;
        #pragma unroll 8
        for (int r = 0; r < 32; ++r) {
          float x = Ap[r], y = Aq[r];
          Ap[r] = cth * x - sth * y;
          Aq[r] = sth * x + cth * y;
          float vx = Vp[r], vy = Vq[r];
          Vp[r] = cth * vx - sth * vy;
          Vq[r] = sth * vx + cth * vy;
        }
      }
      __syncthreads();
    }
  }

  // Singular values = column norms of B*J
  if (tid < 128) {
    float acc = 0.f;
    for (int r = 0; r < 128; ++r) {
      float x = Ac[tid * CP + r];
      acc = fmaf(x, x, acc);
    }
    sv[tid]  = sqrtf(acc);
    idx[tid] = tid;
  }
  __syncthreads();
  if (tid == 0) {                       // descending selection sort (128 elems)
    for (int i = 0; i < 127; ++i) {
      int best = i;
      for (int j = i + 1; j < 128; ++j)
        if (sv[idx[j]] > sv[idx[best]]) best = j;
      int t = idx[i]; idx[i] = idx[best]; idx[best] = t;
    }
  }
  __syncthreads();
  if (tid < 128) sinv[tid] = 1.f / sv[idx[tid]];
  __syncthreads();

  // Layer 0: r1 = relu(Uk @ W1[0])   Uk col j = Vc[:, idx[j]]  (orthonormal)
  for (int e = tid; e < 128 * KK; e += 256) {
    int i = e >> 5, j = e & 31;
    float acc = 0.f;
    #pragma unroll 8
    for (int kk = 0; kk < KK; ++kk)
      acc = fmaf(Vc[idx[kk] * CP + i], w1[kk * KK + j], acc);
    r1[i * RP + j] = fmaxf(acc, 0.f);
  }
  __syncthreads();

  // Layers 1,2: 32x32
  for (int l = 1; l <= 2; ++l) {
    const float* in  = (l == 1) ? r1 : r2;
    float*       out = (l == 1) ? r2 : r1;
    const float* Wl  = w1 + l * KK * KK;
    for (int e = tid; e < 128 * KK; e += 256) {
      int i = e >> 5, j = e & 31;
      float acc = 0.f;
      #pragma unroll 8
      for (int kk = 0; kk < KK; ++kk)
        acc = fmaf(in[i * RP + kk], Wl[kk * KK + j], acc);
      out[i * RP + j] = fmaxf(acc, 0.f);
    }
    __syncthreads();
  }
  // result in r1; scale by Sigma (relu redundant but kept)
  for (int e = tid; e < 128 * KK; e += 256) {
    int i = e >> 5, j = e & 31;
    r1[i * RP + j] = fmaxf(r1[i * RP + j] * sv[idx[j]], 0.f);
  }
  __syncthreads();

  // T = relu(r1 @ W2[1]) : 128x128, overwrite Vc (row-major T[i*CP + j]);
  // Ac must stay alive — it holds the (unnormalized) right singular vectors.
  for (int e = tid; e < 128 * 128; e += 256) {
    int i = e >> 7, j = e & 127;
    float acc = 0.f;
    #pragma unroll 8
    for (int kk = 0; kk < KK; ++kk)
      acc = fmaf(r1[i * RP + kk], w2[KK * WW + kk * WW + j], acc);
    Vc[i * CP + j] = fmaxf(acc, 0.f);
  }
  __syncthreads();

  // rflat = (T @ Vs).reshape(-1), Vs col jj = Ac[:, idx[32+jj]] / s ; bf16 out
  unsigned short* out = rflat + (size_t)mat * KDIM;
  for (int e = tid; e < 128 * 96; e += 256) {
    int i = e / 96, jj = e % 96;
    const float* Trow = Vc + i * CP;
    const float* Vcol = Ac + idx[KK + jj] * CP;
    float acc = 0.f;
    #pragma unroll 8
    for (int r = 0; r < 128; ++r) acc = fmaf(Trow[r], Vcol[r], acc);
    out[i * 96 + jj] = f32_to_bf16(acc * sinv[KK + jj]);
  }
}

// ---------------------------------------------------------------------------
// Kernel 3: Out(256x256) = Rflat(256x12288,bf16) x weightsO^T(12288x256,bf16)
// One 16x16 C-tile per wave32; K-loop of v_wmma_f32_16x16x32_bf16
// ---------------------------------------------------------------------------
__global__ __launch_bounds__(256) void gemm_out_kernel(
    const unsigned short* __restrict__ Rb, const unsigned short* __restrict__ Wb,
    float* __restrict__ out) {
  const int wave = blockIdx.x * 8 + (threadIdx.x >> 5);
  const int lane = threadIdx.x & 31;
  const int mt = wave >> 4;            // 16 M-tiles
  const int nt = wave & 15;            // 16 N-tiles
  const int mrow = mt * 16 + (lane & 15);
  const int nrow = nt * 16 + (lane & 15);
  const int kg   = (lane >> 4) * 8;    // K-half handled by this lane group

  const v4u* Ap = reinterpret_cast<const v4u*>(Rb + (size_t)mrow * KDIM);
  const v4u* Bp = reinterpret_cast<const v4u*>(Wb + (size_t)nrow * KDIM);

  v8f acc = {};
  for (int k = 0; k < KDIM; k += 32) {
    int c0 = (k + kg) >> 3;            // 16B chunk index (8 bf16 per chunk)
    int c1 = (k + 16 + kg) >> 3;
    Frag fa, fb;
    fa.u2[0] = Ap[c0]; fa.u2[1] = Ap[c1];
    fb.u2[0] = Bp[c0]; fb.u2[1] = Bp[c1];
    acc = __builtin_amdgcn_wmma_f32_16x16x32_bf16(
        /*neg_a=*/false, fa.v, /*neg_b=*/false, fb.v,
        /*c_mod=*/(short)0, acc, /*reuse_a=*/false, /*reuse_b=*/false);
  }

  // C layout: VGPR i -> M = i (lanes 0-15) / 8+i (lanes 16-31); N = lane&15
  const int mbase = mt * 16 + (lane >> 4) * 8;
  const int n     = nt * 16 + (lane & 15);
  #pragma unroll
  for (int i = 0; i < 8; ++i)
    out[(size_t)(mbase + i) * AOUT + n] = acc[i];
}

// ---------------------------------------------------------------------------
// Host-side launch
// ---------------------------------------------------------------------------
extern "C" void kernel_launch(void* const* d_in, const int* in_sizes, int n_in,
                              void* d_out, int out_size, void* d_ws, size_t ws_size,
                              hipStream_t stream) {
  (void)in_sizes; (void)n_in; (void)out_size; (void)ws_size;
  const float* obs = (const float*)d_in[0];
  const float* w1  = (const float*)d_in[1];   // (3,32,32)
  const float* w2  = (const float*)d_in[2];   // (2,32,128)
  const float* wO  = (const float*)d_in[3];   // (256,12288)
  const float* c1w = (const float*)d_in[4];
  const float* c1b = (const float*)d_in[5];
  const float* c2w = (const float*)d_in[6];
  const float* c2b = (const float*)d_in[7];
  float* out = (float*)d_out;

  char* ws = (char*)d_ws;
  float*          mats  = (float*)ws;                                  // 16 MB
  unsigned short* rflat = (unsigned short*)(ws + (size_t)16777216);    //  6 MB
  unsigned short* wob   = (unsigned short*)(ws + (size_t)16777216 + 6291456);

  cvt_bf16_kernel<<<(AOUT * KDIM) / 256, 256, 0, stream>>>(wO, wob, AOUT * KDIM);
  build_mats_kernel<<<4096, 256, 0, stream>>>(obs, c1w, c1b, c2w, c2b, mats);

  size_t smem = (size_t)(2 * 128 * CP + 128 + 128 + 2 * 128 * RP) * sizeof(float)
              + 128 * sizeof(int);
  svd_mlp_kernel<<<BB, 256, smem, stream>>>(mats, w1, w2, rflat);

  gemm_out_kernel<<<32, 256, 0, stream>>>(rflat, wob, out);
}
